// InteractionBlock_31559419691084
// MI455X (gfx1250) — compile-verified, matrix operand order checked by reading, not run on hardware
//
#include <hip/hip_runtime.h>

#define N_NODES 10000
#define N_EDGES 640000
#define HIDDEN  128
#define NUM_RBF 64

typedef __attribute__((ext_vector_type(16))) _Float16 v16h;
typedef __attribute__((ext_vector_type(8)))  float    v8f;
typedef __attribute__((ext_vector_type(2)))  _Float16 h2;

// ---------------------------------------------------------------------------
// Load a 16x32 f16 fragment in the CDNA5 WMMA "A" layout from a row-major
// f16 tile (row stride = ld elements).  Per ISA 7.12.2:
//   lane l: row r = l&15, k-offset 8*(l>>4)
//   VGPR v in 0..3 : K = 2v   (+koff), v in 4..7 : K = 16 + 2(v-4) (+koff)
// The same loader serves the B operand of  out = in @ W^T  when applied to
// the row-major weight W[n][k] (n plays the role of the row index).
// ---------------------------------------------------------------------------
__device__ __forceinline__ v16h ld_frag(const _Float16* p, int ld) {
    int lane = threadIdx.x & 31;
    const h2* row = (const h2*)(p + (lane & 15) * ld + ((lane >> 4) << 3));
    v16h f;
#pragma unroll
    for (int v = 0; v < 8; ++v) {
        h2 t = row[(v < 4) ? v : (v + 4)];   // k/2 index: v  or  8+(v-4)
        f[2 * v]     = t.x;
        f[2 * v + 1] = t.y;
    }
    return f;
}

__device__ __forceinline__ v8f wmma16(v16h a, v16h b, v8f c) {
    return __builtin_amdgcn_wmma_f32_16x16x32_f16(
        false, a, false, b, (short)0, c, false, false);
}

// ---------------------------------------------------------------------------
// prep: zero agg accumulator, convert all weight matrices to f16
// ---------------------------------------------------------------------------
__global__ void prep_kernel(const float* w1, const float* w2, const float* l1,
                            const float* l2, const float* bk, float* agg,
                            _Float16* w1h, _Float16* w2h, _Float16* l1h,
                            _Float16* l2h, _Float16* bkh) {
    int i = blockIdx.x * blockDim.x + threadIdx.x;
    if (i < N_NODES * HIDDEN) agg[i] = 0.0f;
    if (i < HIDDEN * NUM_RBF) w1h[i] = (_Float16)w1[i];
    if (i < HIDDEN * HIDDEN) {
        w2h[i] = (_Float16)w2[i];
        l1h[i] = (_Float16)l1[i];
        l2h[i] = (_Float16)l2[i];
        bkh[i] = (_Float16)bk[i];
    }
}

// ---------------------------------------------------------------------------
// h = x @ lin1^T   (10000x128)(128x128), 1 wave = one 16-row tile
// ---------------------------------------------------------------------------
__global__ void __launch_bounds__(256)
lin1_kernel(const float* __restrict__ x, const _Float16* __restrict__ l1h,
            float* __restrict__ h) {
    __shared__ _Float16 s_w[HIDDEN * HIDDEN];
    __shared__ _Float16 s_stage[8][16 * HIDDEN];

    for (int i = threadIdx.x; i < HIDDEN * HIDDEN; i += 256) s_w[i] = l1h[i];
    __syncthreads();

    int wave = threadIdx.x >> 5, lane = threadIdx.x & 31;
    int tile = blockIdx.x * 8 + wave;
    if (tile >= N_NODES / 16) return;
    int r0 = tile * 16;

    _Float16* st = s_stage[wave];
    for (int i = lane; i < 16 * HIDDEN; i += 32)
        st[i] = (_Float16)x[r0 * HIDDEN + i];        // coalesced

    v16h a[4];
#pragma unroll
    for (int kc = 0; kc < 4; ++kc) a[kc] = ld_frag(st + kc * 32, HIDDEN);

    int n_l = lane & 15, mrow = (lane >> 4) * 8;
#pragma unroll
    for (int nt = 0; nt < 8; ++nt) {
        v8f c = {};
#pragma unroll
        for (int kc = 0; kc < 4; ++kc) {
            v16h b = ld_frag(s_w + nt * 16 * HIDDEN + kc * 32, HIDDEN);
            c = wmma16(a[kc], b, c);
        }
        int n = nt * 16 + n_l;
#pragma unroll
        for (int v = 0; v < 8; ++v)
            h[(r0 + mrow + v) * HIDDEN + n] = c[v];
    }
}

// ---------------------------------------------------------------------------
// Fused edge kernel: filter net (2 WMMA GEMMs + tanh), cutoff envelope,
// gather h[src], modulate, atomic scatter-add into agg[dst].
// One wave = 16 edges.  E = 640000 = 5000 blocks * 8 waves * 16 edges.
// ---------------------------------------------------------------------------
__global__ void __launch_bounds__(256)
edge_kernel(const float* __restrict__ eattr, const float* __restrict__ ewt,
            const int* __restrict__ eidx, const _Float16* __restrict__ w1h,
            const float* __restrict__ b1, const _Float16* __restrict__ w2h,
            const float* __restrict__ b2, const float* __restrict__ h,
            float* __restrict__ agg) {
    __shared__ _Float16 s_w1[HIDDEN * NUM_RBF];
    __shared__ _Float16 s_w2[HIDDEN * HIDDEN];
    __shared__ float    s_b1[HIDDEN], s_b2[HIDDEN];
    __shared__ _Float16 s_stage[8][16 * HIDDEN];
    __shared__ float    s_cenv[8][16];
    __shared__ int      s_src[8][16], s_dst[8][16];

    for (int i = threadIdx.x; i < HIDDEN * NUM_RBF; i += 256) s_w1[i] = w1h[i];
    for (int i = threadIdx.x; i < HIDDEN * HIDDEN;  i += 256) s_w2[i] = w2h[i];
    if (threadIdx.x < HIDDEN) {
        s_b1[threadIdx.x] = b1[threadIdx.x];
        s_b2[threadIdx.x] = b2[threadIdx.x];
    }
    __syncthreads();

    int wave = threadIdx.x >> 5, lane = threadIdx.x & 31;
    int e0 = (blockIdx.x * 8 + wave) * 16;

    if (lane < 16) {                                   // per-edge scalars
        int e = e0 + lane;
        float w = ewt[e];
        float C = 0.5f * (__cosf(w * 0.6283185307179586f) + 1.0f);
        s_cenv[wave][lane] = (w < 5.0f) ? C : 0.0f;
        s_src[wave][lane] = eidx[e];
        s_dst[wave][lane] = eidx[N_EDGES + e];
    }

    _Float16* st = s_stage[wave];
    for (int i = lane; i < 16 * NUM_RBF; i += 32)      // stage 16x64 attr tile
        st[i] = (_Float16)eattr[e0 * NUM_RBF + i];     // coalesced

    // GEMM1 A fragments hoisted (frees staging for tanh output)
    v16h a0 = ld_frag(st, NUM_RBF);
    v16h a1 = ld_frag(st + 32, NUM_RBF);

    int n_l = lane & 15, hb = lane >> 4;

    // T1 = tanh(attr @ w1^T + b1)  -> restage as f16 (row stride 128)
#pragma unroll
    for (int nt = 0; nt < 8; ++nt) {
        v8f c = {};
        c = wmma16(a0, ld_frag(s_w1 + nt * 16 * NUM_RBF,      NUM_RBF), c);
        c = wmma16(a1, ld_frag(s_w1 + nt * 16 * NUM_RBF + 32, NUM_RBF), c);
        int n = nt * 16 + n_l;
        float bias = s_b1[n];
#pragma unroll
        for (int v = 0; v < 8; ++v) {
            int m = v + hb * 8;
            st[m * HIDDEN + n] = (_Float16)tanhf(c[v] + bias);
        }
    }

    // A' fragments for GEMM2 (same-wave LDS is in-order: store->load is safe)
    v16h a2[4];
#pragma unroll
    for (int kc = 0; kc < 4; ++kc) a2[kc] = ld_frag(st + kc * 32, HIDDEN);

    // per-v edge scalars (m = v + 8*hb)
    float cenvv[8]; int srcb[8], dstb[8];
#pragma unroll
    for (int v = 0; v < 8; ++v) {
        int m = v + hb * 8;
        cenvv[v] = s_cenv[wave][m];
        srcb[v]  = s_src[wave][m] * HIDDEN;
        dstb[v]  = s_dst[wave][m] * HIDDEN;
    }

    // W = (T1' @ w2^T + b2) * cenv ; msg = h[src]*W ; agg[dst] += msg
#pragma unroll
    for (int nt = 0; nt < 8; ++nt) {
        v8f c = {};
#pragma unroll
        for (int kc = 0; kc < 4; ++kc)
            c = wmma16(a2[kc], ld_frag(s_w2 + nt * 16 * HIDDEN + kc * 32, HIDDEN), c);
        int n = nt * 16 + n_l;
        float bias = s_b2[n];
#pragma unroll
        for (int v = 0; v < 8; ++v) {
            float W   = (c[v] + bias) * cenvv[v];
            float msg = h[srcb[v] + n] * W;          // gather (L2-resident)
            unsafeAtomicAdd(&agg[dstb[v] + n], msg); // global_atomic_add_f32
        }
    }
}

// ---------------------------------------------------------------------------
// out = tanh(agg @ lin2^T + l2b) @ blk^T + bkb   (two chained WMMA GEMMs)
// ---------------------------------------------------------------------------
__global__ void __launch_bounds__(256)
tail_kernel(const float* __restrict__ agg, const _Float16* __restrict__ l2h,
            const float* __restrict__ l2b, const _Float16* __restrict__ bkh,
            const float* __restrict__ bkb, float* __restrict__ out) {
    __shared__ _Float16 s_w2[HIDDEN * HIDDEN];
    __shared__ _Float16 s_wb[HIDDEN * HIDDEN];
    __shared__ float    s_b2[HIDDEN], s_bb[HIDDEN];
    __shared__ _Float16 s_stage[8][16 * HIDDEN];

    for (int i = threadIdx.x; i < HIDDEN * HIDDEN; i += 256) {
        s_w2[i] = l2h[i];
        s_wb[i] = bkh[i];
    }
    if (threadIdx.x < HIDDEN) {
        s_b2[threadIdx.x] = l2b[threadIdx.x];
        s_bb[threadIdx.x] = bkb[threadIdx.x];
    }
    __syncthreads();

    int wave = threadIdx.x >> 5, lane = threadIdx.x & 31;
    int tile = blockIdx.x * 8 + wave;
    if (tile >= N_NODES / 16) return;
    int r0 = tile * 16;

    _Float16* st = s_stage[wave];
    for (int i = lane; i < 16 * HIDDEN; i += 32)
        st[i] = (_Float16)agg[r0 * HIDDEN + i];

    v16h a[4];
#pragma unroll
    for (int kc = 0; kc < 4; ++kc) a[kc] = ld_frag(st + kc * 32, HIDDEN);

    int n_l = lane & 15, hb = lane >> 4;

    // t = tanh(agg @ lin2^T + b)  -> restage
#pragma unroll
    for (int nt = 0; nt < 8; ++nt) {
        v8f c = {};
#pragma unroll
        for (int kc = 0; kc < 4; ++kc)
            c = wmma16(a[kc], ld_frag(s_w2 + nt * 16 * HIDDEN + kc * 32, HIDDEN), c);
        int n = nt * 16 + n_l;
        float bias = s_b2[n];
#pragma unroll
        for (int v = 0; v < 8; ++v) {
            int m = v + hb * 8;
            st[m * HIDDEN + n] = (_Float16)tanhf(c[v] + bias);
        }
    }

    v16h a2[4];
#pragma unroll
    for (int kc = 0; kc < 4; ++kc) a2[kc] = ld_frag(st + kc * 32, HIDDEN);

#pragma unroll
    for (int nt = 0; nt < 8; ++nt) {
        v8f c = {};
#pragma unroll
        for (int kc = 0; kc < 4; ++kc)
            c = wmma16(a2[kc], ld_frag(s_wb + nt * 16 * HIDDEN + kc * 32, HIDDEN), c);
        int n = nt * 16 + n_l;
        float bias = s_bb[n];
#pragma unroll
        for (int v = 0; v < 8; ++v)
            out[(r0 + (lane >> 4) * 8 + v) * HIDDEN + n] = c[v] + bias;
    }
}

// ---------------------------------------------------------------------------
extern "C" void kernel_launch(void* const* d_in, const int* in_sizes, int n_in,
                              void* d_out, int out_size, void* d_ws, size_t ws_size,
                              hipStream_t stream) {
    const float* x     = (const float*)d_in[0];
    const int*   eidx  = (const int*)d_in[1];
    const float* ewt   = (const float*)d_in[2];
    const float* eattr = (const float*)d_in[3];
    // d_in[4] atom_types, d_in[5] seq_neighs: unused by the reference
    const float* w1  = (const float*)d_in[6];
    const float* b1  = (const float*)d_in[7];
    const float* w2  = (const float*)d_in[8];
    const float* b2  = (const float*)d_in[9];
    const float* l1  = (const float*)d_in[10];
    const float* l2  = (const float*)d_in[11];
    const float* l2b = (const float*)d_in[12];
    const float* bk  = (const float*)d_in[13];
    const float* bkb = (const float*)d_in[14];
    float* out = (float*)d_out;

    char* ws = (char*)d_ws;
    float*    agg  = (float*)ws;                               // 5,120,000 B
    float*    hbuf = (float*)(ws + 5120000);                   // 5,120,000 B
    _Float16* w1h  = (_Float16*)(ws + 10240000);               // 16 KB
    _Float16* w2h  = (_Float16*)(ws + 10240000 + 16384);       // 32 KB
    _Float16* l1h  = (_Float16*)(ws + 10240000 + 49152);
    _Float16* l2h  = (_Float16*)(ws + 10240000 + 81920);
    _Float16* bkh  = (_Float16*)(ws + 10240000 + 114688);

    prep_kernel<<<(N_NODES * HIDDEN + 255) / 256, 256, 0, stream>>>(
        w1, w2, l1, l2, bk, agg, w1h, w2h, l1h, l2h, bkh);

    lin1_kernel<<<(N_NODES / 16 + 7) / 8, 256, 0, stream>>>(x, l1h, hbuf);

    edge_kernel<<<N_EDGES / (8 * 16), 256, 0, stream>>>(
        eattr, ewt, eidx, w1h, b1, w2h, b2, hbuf, agg);

    tail_kernel<<<(N_NODES / 16 + 7) / 8, 256, 0, stream>>>(
        agg, l2h, l2b, bkh, bkb, out);
}
